// GCNOLD_32719060861208
// MI455X (gfx1250) — compile-verified, matrix operand order checked by reading, not run on hardware
//
#include <hip/hip_runtime.h>
#include <cstddef>

// ---------------------------------------------------------------------------
// GNN pipeline for MI455X (gfx1250, wave32).
// Dense matmuls: v_wmma_f32_16x16x32_f16, 32x32 output tile per wave
// (2x2 register tiling, pre-packed f16 weights in B-fragment lane layout).
// Graph aggregation: native global_atomic_add_f32 (L2-resident tensors).
// ---------------------------------------------------------------------------

typedef __attribute__((ext_vector_type(16))) _Float16 v16h;
typedef __attribute__((ext_vector_type(8)))  float    v8f;

#define ACT_NONE 0
#define ACT_RELU 1
#define ACT_ELU  2

template <int ACT>
__device__ __forceinline__ float act_apply(float v) {
    if (ACT == ACT_RELU) return fmaxf(v, 0.f);
    if (ACT == ACT_ELU)  return fmaxf(v, 0.f) + __expf(fminf(v, 0.f)) - 1.f;
    return v;
}

__device__ __forceinline__ void atom_add_f32(float* p, float v) {
    __hip_atomic_fetch_add(p, v, __ATOMIC_RELAXED, __HIP_MEMORY_SCOPE_AGENT);
}

// ---------------------------------------------------------------------------
// Weight pre-pack: W[Fin x Fout] f32 -> P, f16 fragments in WMMA B lane order.
// Fragment (t = 16-col tile, c = 32-K chunk): lane l holds 16 halves where
//   k = c*32 + 8*(l/16) + (i<8 ? i : i+8),  col = t*16 + (l%16)
// P index: ((t*nchunks + c)*32 + lane) * 16 halves  (32B per lane, contiguous)
// ---------------------------------------------------------------------------
__global__ void __launch_bounds__(32)
packW_kernel(const float* __restrict__ W, _Float16* __restrict__ P, int Fout) {
    const int t = blockIdx.x, c = blockIdx.y, nchunks = gridDim.y;
    const int lane = threadIdx.x;
    const int half = lane >> 4, l16 = lane & 15;
    const int col = t * 16 + l16;
    v16h v;
#pragma unroll
    for (int i = 0; i < 16; ++i) {
        const int k = c * 32 + half * 8 + (i < 8 ? i : i + 8);
        v[i] = (_Float16)W[(size_t)k * Fout + col];
    }
    ((v16h*)P)[((size_t)t * nchunks + c) * 32 + lane] = v;
}

// ---------------------------------------------------------------------------
// WMMA GEMM:  Y[Rows x Fout] = act( X[Rows x FIN] @ W + bias ),  W pre-packed.
// One wave computes a 32x32 output tile (2x2 grid of 16x16 WMMA tiles).
// A loads are unconditional (row pointer clamped): tail rows compute garbage
// that is simply not stored -> no exec-mask branching in the hot loop.
// ---------------------------------------------------------------------------
template <int FIN, int ACT>
__global__ void __launch_bounds__(32)
gemm_wmma_kernel(const float* __restrict__ X, const _Float16* __restrict__ packB,
                 const float* __restrict__ bias, float* __restrict__ Y,
                 int Rows, int Fout)
{
    constexpr int NCH = FIN / 32;
    const int lane = threadIdx.x;
    const int half = lane >> 4;
    const int l16  = lane & 15;
    const int tm   = blockIdx.x;  // 32-row tile
    const int tn   = blockIdx.y;  // 32-col tile
    const int row0 = tm * 32 + l16;
    const int row1 = row0 + 16;
    const int col0 = tn * 32 + l16;
    const int col1 = col0 + 16;

    const float* xr0 = X + (size_t)(row0 < Rows ? row0 : Rows - 1) * FIN;
    const float* xr1 = X + (size_t)(row1 < Rows ? row1 : Rows - 1) * FIN;
    const v16h* pb0 = (const v16h*)packB + ((size_t)(tn * 2 + 0) * NCH) * 32 + lane;
    const v16h* pb1 = (const v16h*)packB + ((size_t)(tn * 2 + 1) * NCH) * 32 + lane;

    v8f acc00, acc01, acc10, acc11;
    const float bi0 = bias ? bias[col0] : 0.f;
    const float bi1 = bias ? bias[col1] : 0.f;
#pragma unroll
    for (int r = 0; r < 8; ++r) {
        acc00[r] = bi0; acc01[r] = bi1;
        acc10[r] = bi0; acc11[r] = bi1;
    }

#pragma unroll
    for (int c = 0; c < NCH; ++c) {
        const int kbase = c * 32 + half * 8;
        v16h a0, a1;
#pragma unroll
        for (int i = 0; i < 8; ++i) {          // contiguous 8-float runs -> b128
            a0[i]     = (_Float16)xr0[kbase + i];
            a0[i + 8] = (_Float16)xr0[kbase + 16 + i];
            a1[i]     = (_Float16)xr1[kbase + i];
            a1[i + 8] = (_Float16)xr1[kbase + 16 + i];
        }
        const v16h b0 = pb0[c * 32];           // one 32B packed fragment
        const v16h b1 = pb1[c * 32];
        acc00 = __builtin_amdgcn_wmma_f32_16x16x32_f16(false, a0, false, b0, (short)0, acc00, false, false);
        acc01 = __builtin_amdgcn_wmma_f32_16x16x32_f16(false, a0, false, b1, (short)0, acc01, false, false);
        acc10 = __builtin_amdgcn_wmma_f32_16x16x32_f16(false, a1, false, b0, (short)0, acc10, false, false);
        acc11 = __builtin_amdgcn_wmma_f32_16x16x32_f16(false, a1, false, b1, (short)0, acc11, false, false);
    }

    const int mbase = tm * 32 + half * 8;
    if (tm * 32 + 32 <= Rows) {                // uniform: branch-free store path
#pragma unroll
        for (int r = 0; r < 8; ++r) {
            float* y0 = Y + (size_t)(mbase + r) * Fout;
            float* y1 = Y + (size_t)(mbase + 16 + r) * Fout;
            y0[col0] = act_apply<ACT>(acc00[r]);
            y0[col1] = act_apply<ACT>(acc01[r]);
            y1[col0] = act_apply<ACT>(acc10[r]);
            y1[col1] = act_apply<ACT>(acc11[r]);
        }
    } else {                                   // tail row-tile only
#pragma unroll
        for (int r = 0; r < 8; ++r) {
            if (mbase + r < Rows) {
                float* y0 = Y + (size_t)(mbase + r) * Fout;
                y0[col0] = act_apply<ACT>(acc00[r]);
                y0[col1] = act_apply<ACT>(acc01[r]);
            }
            if (mbase + 16 + r < Rows) {
                float* y1 = Y + (size_t)(mbase + 16 + r) * Fout;
                y1[col0] = act_apply<ACT>(acc10[r]);
                y1[col1] = act_apply<ACT>(acc11[r]);
            }
        }
    }
}

// ---------------------------------------------------------------------------
// Small helpers
// ---------------------------------------------------------------------------
__global__ void zero_kernel(float* __restrict__ p, long n) {
    long i = (long)blockIdx.x * blockDim.x + threadIdx.x;
    if (i < n) p[i] = 0.f;
}

__global__ void deg_kernel(const int* __restrict__ dst, float* __restrict__ deg, int E) {
    int e = blockIdx.x * blockDim.x + threadIdx.x;
    if (e < E) atom_add_f32(&deg[dst[e]], 1.f);
}

__global__ void dinv_kernel(float* __restrict__ d, int n) {
    int i = blockIdx.x * blockDim.x + threadIdx.x;
    if (i < n) d[i] = rsqrtf(d[i] + 1.f);   // deg with self loop, ^-0.5
}

__global__ void ecoef_kernel(const int* __restrict__ src, const int* __restrict__ dst,
                             const float* __restrict__ dinv, float* __restrict__ coef, int E) {
    int e = blockIdx.x * blockDim.x + threadIdx.x;
    if (e < E) coef[e] = dinv[src[e]] * dinv[dst[e]];
}

// agg = dinv^2 * hw + bias  (self-loop term + bias, also serves as zero-init)
__global__ void self_init_kernel(const float* __restrict__ hw, const float* __restrict__ dinv,
                                 const float* __restrict__ bias, float* __restrict__ agg,
                                 int n, int F) {
    long idx = (long)blockIdx.x * blockDim.x + threadIdx.x;
    if (idx >= (long)n * F) return;
    int i = (int)(idx / F), f = (int)(idx % F);
    float dv = dinv[i];
    agg[idx] = dv * dv * hw[idx] + bias[f];
}

// agg[dst] += coef[e] * hw[src], one thread per (edge, float4 chunk)
__global__ void __launch_bounds__(256)
scatter_kernel(const float* __restrict__ hw, const int* __restrict__ src,
               const int* __restrict__ dst, const float* __restrict__ coef,
               float* __restrict__ agg, int E, int F) {
    const int F4 = F >> 2;
    long idx = (long)blockIdx.x * blockDim.x + threadIdx.x;
    if (idx >= (long)E * F4) return;
    int e = (int)(idx / F4), c = (int)(idx % F4);
    int s = src[e], d = dst[e];
    float cf = coef[e];
    const float4 v = ((const float4*)(hw + (size_t)s * F))[c];
    float* ap = agg + (size_t)d * F + (size_t)c * 4;
    atom_add_f32(ap + 0, v.x * cf);
    atom_add_f32(ap + 1, v.y * cf);
    atom_add_f32(ap + 2, v.z * cf);
    atom_add_f32(ap + 3, v.w * cf);
}

template <int ACT>
__global__ void act_copy_kernel(const float* __restrict__ in, float* __restrict__ out, long n) {
    long i = (long)blockIdx.x * blockDim.x + threadIdx.x;
    if (i < n) out[i] = act_apply<ACT>(in[i]);
}

// per-feature mean / rsqrt(var+eps) over n rows; one block per feature
__global__ void __launch_bounds__(256)
instnorm_stats_kernel(const float* __restrict__ H, float* __restrict__ stats, int n, int F) {
    __shared__ float ssum[256], ssq[256];
    const int f = blockIdx.x;
    float s = 0.f, q = 0.f;
    for (int i = threadIdx.x; i < n; i += 256) {
        float v = H[(size_t)i * F + f];
        s += v; q += v * v;
    }
    ssum[threadIdx.x] = s; ssq[threadIdx.x] = q;
    __syncthreads();
    for (int st = 128; st > 0; st >>= 1) {
        if (threadIdx.x < st) {
            ssum[threadIdx.x] += ssum[threadIdx.x + st];
            ssq[threadIdx.x]  += ssq[threadIdx.x + st];
        }
        __syncthreads();
    }
    if (threadIdx.x == 0) {
        float mu  = ssum[0] / (float)n;
        float var = ssq[0] / (float)n - mu * mu;
        stats[f * 2 + 0] = mu;
        stats[f * 2 + 1] = rsqrtf(var + 1e-5f);
    }
}

__global__ void cnt_kernel(const int* __restrict__ cluster, float* __restrict__ cnt, int n) {
    int i = blockIdx.x * blockDim.x + threadIdx.x;
    if (i < n) atom_add_f32(&cnt[cluster[i]], 1.f);
}

// zsum[cluster[i]] += (H - mu) * rsig  (instance-norm fused into pooling)
__global__ void pool_kernel(const float* __restrict__ H, const float* __restrict__ stats,
                            const int* __restrict__ cluster, float* __restrict__ zsum,
                            int n, int F) {
    long idx = (long)blockIdx.x * blockDim.x + threadIdx.x;
    if (idx >= (long)n * F) return;
    int i = (int)(idx / F), f = (int)(idx % F);
    float y = (H[idx] - stats[f * 2]) * stats[f * 2 + 1];
    atom_add_f32(&zsum[(size_t)cluster[i] * F + f], y);
}

__global__ void pool_div_kernel(float* __restrict__ z, const float* __restrict__ cnt,
                                int Cn, int F) {
    int idx = blockIdx.x * blockDim.x + threadIdx.x;
    if (idx >= Cn * F) return;
    z[idx] /= fmaxf(cnt[idx / F], 1.f);
}

__global__ void gather_kernel(const float* __restrict__ z, const int* __restrict__ cluster,
                              float* __restrict__ up, int n, int F) {
    long idx = (long)blockIdx.x * blockDim.x + threadIdx.x;
    if (idx >= (long)n * F) return;
    int i = (int)(idx / F), f = (int)(idx % F);
    up[idx] = z[(size_t)cluster[i] * F + f];
}

// naive linear for shapes WMMA can't tile (6->32, 32->3)
template <int ACT>
__global__ void lin_naive_kernel(const float* __restrict__ X, const float* __restrict__ W,
                                 const float* __restrict__ bias, float* __restrict__ Y,
                                 int Rows, int Fin, int Fout) {
    long idx = (long)blockIdx.x * blockDim.x + threadIdx.x;
    if (idx >= (long)Rows * Fout) return;
    int i = (int)(idx / Fout), f = (int)(idx % Fout);
    float s = bias[f];
    for (int k = 0; k < Fin; ++k) s += X[(size_t)i * Fin + k] * W[(size_t)k * Fout + f];
    Y[idx] = act_apply<ACT>(s);
}

// ---------------------------------------------------------------------------
// Host orchestration
// ---------------------------------------------------------------------------
static inline dim3 cover(long n, int b) { return dim3((unsigned)((n + b - 1) / b)); }

extern "C" void kernel_launch(void* const* d_in, const int* in_sizes, int n_in,
                              void* d_out, int out_size, void* d_ws, size_t ws_size,
                              hipStream_t stream) {
    (void)n_in; (void)out_size; (void)ws_size;
    const float* x      = (const float*)d_in[0];
    const int*   adj    = (const int*)d_in[1];
    const int*   clus   = (const int*)d_in[2];
    const int*   c_adj  = (const int*)d_in[5];
    const int N  = in_sizes[0] / 6;
    const int E  = in_sizes[1] / 2;
    const int C  = in_sizes[4];           // cluster_parent has C entries
    const int EC = in_sizes[5] / 2;
    const int* src  = adj;        const int* dst  = adj + E;
    const int* csrc = c_adj;      const int* cdst = c_adj + EC;

    // params flattened depth-first in setup_inputs() insertion order: W then b
    enum { FC1=0,FC2,FC3,G1,G2,G3,G4,G5,FT1,FT2,FT3,
           G6,G7,G8,G9,G10,G11,G12,G13,G14,G15,OF1,OF2,OF3,OF4 };
    auto Wp = [&](int l) { return (const float*)d_in[6 + 2 * l]; };
    auto Bp = [&](int l) { return (const float*)d_in[7 + 2 * l]; };

    // workspace carve (256B aligned)
    char* wsp = (char*)d_ws;
    auto carve = [&](size_t bytes) -> void* {
        void* p = (void*)wsp;
        wsp += ((bytes + 255) / 256) * 256;
        return p;
    };
    float* dinv   = (float*)carve((size_t)N * 4);
    float* cdinv  = (float*)carve((size_t)C * 4);
    float* ecoef  = (float*)carve((size_t)E * 4);
    float* ecoefC = (float*)carve((size_t)EC * 4);
    float* stats  = (float*)carve(256 * 4);
    float* cnt    = (float*)carve((size_t)C * 4);
    _Float16* wpack = (_Float16*)carve((size_t)256 * 256 * 2);   // packed weights
    float* hA     = (float*)carve((size_t)N * 128 * 4);
    float* hB     = (float*)carve((size_t)N * 128 * 4);
    float* hC     = (float*)carve((size_t)N * 128 * 4);
    float* zA     = (float*)carve((size_t)C * 256 * 4);
    float* zB     = (float*)carve((size_t)C * 256 * 4);
    float* zC     = (float*)carve((size_t)C * 256 * 4);

    auto gemm = [&](const float* X, const float* W, const float* b, float* Y,
                    int rows, int fin, int fout, int act) {
        packW_kernel<<<dim3(fout / 16, fin / 32), 32, 0, stream>>>(W, wpack, fout);
        dim3 g((rows + 31) / 32, fout / 32);
#define GEMM_CASE(FIN)                                                                          \
        do {                                                                                    \
            if (act == ACT_NONE)                                                                \
                gemm_wmma_kernel<FIN, ACT_NONE><<<g, 32, 0, stream>>>(X, wpack, b, Y, rows, fout); \
            else if (act == ACT_RELU)                                                           \
                gemm_wmma_kernel<FIN, ACT_RELU><<<g, 32, 0, stream>>>(X, wpack, b, Y, rows, fout); \
            else                                                                                \
                gemm_wmma_kernel<FIN, ACT_ELU><<<g, 32, 0, stream>>>(X, wpack, b, Y, rows, fout);  \
        } while (0)
        switch (fin) {
            case 32:  GEMM_CASE(32);  break;
            case 64:  GEMM_CASE(64);  break;
            case 128: GEMM_CASE(128); break;
            case 256: GEMM_CASE(256); break;
        }
#undef GEMM_CASE
    };

    // GCN layer: cur <- act( D^-1/2 (A+I) D^-1/2 (cur W) + b )
    auto gcn = [&](float* cur, int layer, int rows, int fin,
                   const int* s, const int* d, const float* cf, const float* dv,
                   int ne, int act, float* hw, float* agg) {
        gemm(cur, Wp(layer), nullptr, hw, rows, fin, 128, ACT_NONE);
        long nF = (long)rows * 128;
        self_init_kernel<<<cover(nF, 256), 256, 0, stream>>>(hw, dv, Bp(layer), agg, rows, 128);
        scatter_kernel<<<cover((long)ne * 32, 256), 256, 0, stream>>>(hw, s, d, cf, agg, ne, 128);
        if (act == ACT_ELU)
            act_copy_kernel<ACT_ELU><<<cover(nF, 256), 256, 0, stream>>>(agg, cur, nF);
        else
            act_copy_kernel<ACT_NONE><<<cover(nF, 256), 256, 0, stream>>>(agg, cur, nF);
    };

    // --- degree / edge coefficients (reused by all layers) ---
    zero_kernel<<<cover(N, 256), 256, 0, stream>>>(dinv, N);
    deg_kernel<<<cover(E, 256), 256, 0, stream>>>(dst, dinv, E);
    dinv_kernel<<<cover(N, 256), 256, 0, stream>>>(dinv, N);
    ecoef_kernel<<<cover(E, 256), 256, 0, stream>>>(src, dst, dinv, ecoef, E);
    zero_kernel<<<cover(C, 256), 256, 0, stream>>>(cdinv, C);
    deg_kernel<<<cover(EC, 256), 256, 0, stream>>>(cdst, cdinv, EC);
    dinv_kernel<<<cover(C, 256), 256, 0, stream>>>(cdinv, C);
    ecoef_kernel<<<cover(EC, 256), 256, 0, stream>>>(csrc, cdst, cdinv, ecoefC, EC);

    // --- FC stack: 6 -> 32 -> 64 -> 128 ---
    lin_naive_kernel<ACT_RELU><<<cover((long)N * 32, 256), 256, 0, stream>>>(x, Wp(FC1), Bp(FC1), hB, N, 6, 32);
    gemm(hB, Wp(FC2), Bp(FC2), hC, N, 32, 64, ACT_RELU);
    gemm(hC, Wp(FC3), Bp(FC3), hA, N, 64, 128, ACT_NONE);

    // --- GCN block 1 (fine graph) ---
    gcn(hA, G1, N, 128, src, dst, ecoef, dinv, E, ACT_ELU, hB, hC);
    gcn(hA, G2, N, 128, src, dst, ecoef, dinv, E, ACT_ELU, hB, hC);
    gcn(hA, G3, N, 128, src, dst, ecoef, dinv, E, ACT_ELU, hB, hC);
    gcn(hA, G4, N, 128, src, dst, ecoef, dinv, E, ACT_ELU, hB, hC);
    gcn(hA, G5, N, 128, src, dst, ecoef, dinv, E, ACT_NONE, hB, hC);

    // --- InstanceNorm + avg_pool into clusters ---
    instnorm_stats_kernel<<<128, 256, 0, stream>>>(hA, stats, N, 128);
    zero_kernel<<<cover(C, 256), 256, 0, stream>>>(cnt, C);
    cnt_kernel<<<cover(N, 256), 256, 0, stream>>>(clus, cnt, N);
    zero_kernel<<<cover((long)C * 128, 256), 256, 0, stream>>>(zA, (long)C * 128);
    pool_kernel<<<cover((long)N * 128, 256), 256, 0, stream>>>(hA, stats, clus, zA, N, 128);
    pool_div_kernel<<<cover((long)C * 128, 256), 256, 0, stream>>>(zA, cnt, C, 128);

    // --- coarse FC stack: 128 -> 256 -> 256 -> 256 ---
    gemm(zA, Wp(FT1), Bp(FT1), zB, C, 128, 256, ACT_ELU);
    gemm(zB, Wp(FT2), Bp(FT2), zC, C, 256, 256, ACT_ELU);
    gemm(zC, Wp(FT3), Bp(FT3), zA, C, 256, 256, ACT_NONE);

    // --- GCN block 2 (coarse graph) ---
    gcn(zA, G6,  C, 256, csrc, cdst, ecoefC, cdinv, EC, ACT_ELU, zB, zC);
    gcn(zA, G7,  C, 128, csrc, cdst, ecoefC, cdinv, EC, ACT_ELU, zB, zC);
    gcn(zA, G8,  C, 128, csrc, cdst, ecoefC, cdinv, EC, ACT_ELU, zB, zC);
    gcn(zA, G9,  C, 128, csrc, cdst, ecoefC, cdinv, EC, ACT_ELU, zB, zC);
    gcn(zA, G10, C, 128, csrc, cdst, ecoefC, cdinv, EC, ACT_NONE, zB, zC);

    // --- upsample: up[i] = z[cluster[i]] ---
    gather_kernel<<<cover((long)N * 128, 256), 256, 0, stream>>>(zA, clus, hA, N, 128);

    // --- GCN block 3 (fine graph) ---
    gcn(hA, G11, N, 128, src, dst, ecoef, dinv, E, ACT_ELU, hB, hC);
    gcn(hA, G12, N, 128, src, dst, ecoef, dinv, E, ACT_ELU, hB, hC);
    gcn(hA, G13, N, 128, src, dst, ecoef, dinv, E, ACT_ELU, hB, hC);
    gcn(hA, G14, N, 128, src, dst, ecoef, dinv, E, ACT_ELU, hB, hC);
    gcn(hA, G15, N, 128, src, dst, ecoef, dinv, E, ACT_NONE, hB, hC);

    // --- output MLP: 128 -> 128 -> 64 -> 32 -> 3 ---
    gemm(hA, Wp(OF1), Bp(OF1), hB, N, 128, 128, ACT_ELU);
    gemm(hB, Wp(OF2), Bp(OF2), hC, N, 128, 64, ACT_ELU);
    gemm(hC, Wp(OF3), Bp(OF3), hB, N, 64, 32, ACT_ELU);
    lin_naive_kernel<ACT_NONE><<<cover((long)N * 3, 256), 256, 0, stream>>>(hB, Wp(OF4), Bp(OF4),
                                                                            (float*)d_out, N, 32, 3);
}
